// SlidingWindowAttention_83485574299656
// MI455X (gfx1250) — matrix operand order repaired and checked by reading, not compiled, not run on hardware
//
#include <hip/hip_runtime.h>
#include <math.h>

// ---------------------------------------------------------------------------
// Sliding-window attention for MI455X (gfx1250), bf16 WMMA pipeline.
// B=2, S=2048, D=1024, H=16, hd=64, window=128 (half=64).
// ---------------------------------------------------------------------------

typedef __bf16 bf16;
typedef bf16  bf16x4  __attribute__((ext_vector_type(4)));
typedef bf16  bf16x8  __attribute__((ext_vector_type(8)));
typedef bf16  bf16x16 __attribute__((ext_vector_type(16)));
typedef float f32x4   __attribute__((ext_vector_type(4)));
typedef float f32x8   __attribute__((ext_vector_type(8)));

#define DEV static __device__ __forceinline__

constexpr int   Bn    = 2;
constexpr int   Sn    = 2048;
constexpr int   Dn    = 1024;
constexpr int   Hn    = 16;
constexpr int   HDn   = 64;
constexpr int   D3    = 3 * Dn;          // 3072
constexpr int   SV    = Sn + 32;         // padded Vt row (absorbs last key chunk)
constexpr float SCALE = 0.125f;          // hd^-0.5 = 1/8
constexpr int   HALFW = 64;              // window/2

DEV f32x8 wmma_bf16(bf16x16 a, bf16x16 b, f32x8 c) {
  // D = A(16x32) * B(32x16) + C, f32 accumulate
  return __builtin_amdgcn_wmma_f32_16x16x32_bf16(false, a, false, b,
                                                 (short)0, c, false, false);
}

// A-matrix 16x32 bf16 tile from a row-major [*, ld] bf16 matrix.
// Lane l: row m = l&15; kb = (l<16)?0:16; elems 0..15 = K=kb..kb+15
// (contiguous K order, matching load_b_tile).
DEV bf16x16 load_a_tile(const bf16* __restrict__ base, int row0, int ld,
                        int k0, int lane) {
  const int m  = lane & 15;
  const int kb = (lane < 16) ? 0 : 16;
  const bf16* p = base + (size_t)(row0 + m) * ld + (k0 + kb);
  bf16x8 lo = *(const bf16x8*)(p);
  bf16x8 hi = *(const bf16x8*)(p + 8);
  return __builtin_shufflevector(lo, hi, 0, 1, 2, 3, 4, 5, 6, 7,
                                 8, 9, 10, 11, 12, 13, 14, 15);
}

// B-matrix 32x16 bf16 tile: lane l owns column n=l&15; needs B[kb2..kb2+15][n]
// K-contiguous -> one 32-byte contiguous load from the per-lane row pointer.
DEV bf16x16 load_b_tile(const bf16* __restrict__ rowp, int k0, int lane) {
  const int kb2 = (lane < 16) ? 0 : 16;
  return *(const bf16x16*)(rowp + k0 + kb2);
}

DEV int imin(int a, int b) { return a < b ? a : b; }

// ---------------------------------------------------------------------------
// Kernel 1: f32 -> bf16 cast, 4 elements/thread (n is a multiple of 1024)
// ---------------------------------------------------------------------------
__global__ void swa_cast_bf16(const float* __restrict__ src,
                              bf16* __restrict__ dst, int n4) {
  for (int i = blockIdx.x * blockDim.x + threadIdx.x; i < n4;
       i += gridDim.x * blockDim.x) {
    f32x4 v = *(const f32x4*)(src + 4 * (size_t)i);
    bf16x4 o = {(bf16)v.x, (bf16)v.y, (bf16)v.z, (bf16)v.w};
    *(bf16x4*)(dst + 4 * (size_t)i) = o;
  }
}

// ---------------------------------------------------------------------------
// Kernel 2: QKV GEMM  qkv[n,c] = sum_k x[n,k] * W[c,k] + b[c]
// One wave -> 32 rows x 64 cols (2 A-tiles x 4 B-tiles, 8 WMMAs / K-step,
// 1.5 b128 loads per WMMA).  Epilogue scatters into:
//   Q  [B,H,S,64]   K [B,H,S,64]   Vt [B,H,64,S+32]   (all bf16)
// ---------------------------------------------------------------------------
__global__ __launch_bounds__(256) void swa_qkv_gemm(
    const bf16* __restrict__ xb, const bf16* __restrict__ wb,
    const float* __restrict__ bias, bf16* __restrict__ qO,
    bf16* __restrict__ kO, bf16* __restrict__ vtO) {
  const int lane = threadIdx.x & 31;
  const int unit = blockIdx.x * (blockDim.x >> 5) + (threadIdx.x >> 5);
  constexpr int COLG = D3 / 64;  // 48
  const int rt = unit / COLG;    // 32-row tile index (Bn*Sn/32 = 128)
  const int cg = unit % COLG;    // 64-col group
  const int n  = lane & 15;
  const int g  = lane >> 4;

  f32x8 acc[2][4] = {};
  for (int k0 = 0; k0 < Dn; k0 += 32) {
    bf16x16 a0 = load_a_tile(xb, rt * 32, Dn, k0, lane);
    bf16x16 a1 = load_a_tile(xb, rt * 32 + 16, Dn, k0, lane);
#pragma unroll
    for (int t = 0; t < 4; ++t) {
      const int c = cg * 64 + t * 16 + n;
      bf16x16 b = load_b_tile(wb + (size_t)c * Dn, k0, lane);
      acc[0][t] = wmma_bf16(a0, b, acc[0][t]);
      acc[1][t] = wmma_bf16(a1, b, acc[1][t]);
    }
  }

#pragma unroll
  for (int t = 0; t < 4; ++t) {
    const int c     = cg * 64 + t * 16 + n;
    const float bv  = bias[c];
    const int which = c >> 10;          // uniform within 16-col tile
    const int rem   = c & 1023;
    const int h     = rem >> 6;
    const int d     = rem & 63;
#pragma unroll
    for (int half = 0; half < 2; ++half) {
#pragma unroll
      for (int r = 0; r < 8; ++r) {
        const int grow = rt * 32 + half * 16 + r + 8 * g;  // in [0, B*S)
        const int bb   = grow >> 11;                       // / S
        const int sq   = grow & (Sn - 1);
        const bf16 v   = (bf16)(acc[half][t][r] + bv);
        if (which == 0)
          qO[(((size_t)(bb * Hn + h)) * Sn + sq) * HDn + d] = v;
        else if (which == 1)
          kO[(((size_t)(bb * Hn + h)) * Sn + sq) * HDn + d] = v;
        else
          vtO[(((size_t)(bb * Hn + h)) * HDn + d) * SV + sq] = v;
      }
    }
  }
}

// ---------------------------------------------------------------------------
// Kernel 3: banded attention, flash-style online softmax.
// grid = (S/16, B*H), block = 32 (one wave).
// ---------------------------------------------------------------------------
__global__ __launch_bounds__(32) void swa_attention(
    const bf16* __restrict__ qb, const bf16* __restrict__ kb,
    const bf16* __restrict__ vtb, bf16* __restrict__ ctx) {
  __shared__ bf16 lds_p[16 * 32];  // P tile staging (C-layout -> A-layout)

  const int lane = threadIdx.x & 31;
  const int q0   = blockIdx.x * 16;
  const int bh   = blockIdx.y;
  const int n    = lane & 15;
  const int g    = lane >> 4;
  const int kbA  = (lane < 16) ? 0 : 16;
  const int kb2  = (lane < 16) ? 0 : 16;

  const bf16* qp = qb + (size_t)bh * Sn * HDn;
  const bf16* kp = kb + (size_t)bh * Sn * HDn;
  const bf16* vp = vtb + (size_t)bh * HDn * SV;

  // Q tile as two A-matrices covering hd = 0..31, 32..63
  const bf16x16 aq0 = load_a_tile(qp, q0, HDn, 0, lane);
  const bf16x16 aq1 = load_a_tile(qp, q0, HDn, 32, lane);

  f32x8 o[4] = {};
  float m[8], l[8];
#pragma unroll
  for (int r = 0; r < 8; ++r) { m[r] = -__builtin_inff(); l[r] = 0.0f; }

  const int kc0   = (q0 <= HALFW) ? 0 : ((q0 - HALFW) & ~31);
  const int kcend = imin(Sn, q0 + 16 + HALFW);

  for (int kc = kc0; kc < kcend; kc += 32) {
    // ---- scores: S = Q K^T (16q x 32k), two 16-key half tiles ----
    f32x8 s[2];
#pragma unroll
    for (int t = 0; t < 2; ++t) {
      const int keyc = imin(kc + t * 16 + n, Sn - 1);  // clamp load only
      const bf16* krow = kp + (size_t)keyc * HDn;
      f32x8 z = {};
      z = wmma_bf16(aq0, load_b_tile(krow, 0, lane), z);
      z = wmma_bf16(aq1, load_b_tile(krow, 32, lane), z);
      s[t] = z;
    }

    // ---- mask + online softmax (row r+8g lives in matching lanes) ----
#pragma unroll
    for (int r = 0; r < 8; ++r) {
      const int qrow = q0 + r + 8 * g;
      const int k0v  = kc + n;
      const int k1v  = kc + 16 + n;
      const bool ok0 = (k0v < Sn) & (k0v >= qrow - HALFW) & (k0v <= qrow + HALFW);
      const bool ok1 = (k1v < Sn) & (k1v >= qrow - HALFW) & (k1v <= qrow + HALFW);
      float v0 = ok0 ? s[0][r] * SCALE : -__builtin_inff();
      float v1 = ok1 ? s[1][r] * SCALE : -__builtin_inff();

      float rm = fmaxf(v0, v1);
      rm = fmaxf(rm, __shfl_xor(rm, 1, 32));
      rm = fmaxf(rm, __shfl_xor(rm, 2, 32));
      rm = fmaxf(rm, __shfl_xor(rm, 4, 32));
      rm = fmaxf(rm, __shfl_xor(rm, 8, 32));

      const float mn    = fmaxf(m[r], rm);
      const float alpha = __expf(m[r] - mn);
      const float p0    = __expf(v0 - mn);
      const float p1    = __expf(v1 - mn);
      float rs = p0 + p1;
      rs += __shfl_xor(rs, 1, 32);
      rs += __shfl_xor(rs, 2, 32);
      rs += __shfl_xor(rs, 4, 32);
      rs += __shfl_xor(rs, 8, 32);

      l[r] = l[r] * alpha + rs;
      m[r] = mn;
#pragma unroll
      for (int t4 = 0; t4 < 4; ++t4) o[t4][r] *= alpha;

      const int qloc = r + 8 * g;
      lds_p[qloc * 32 + n]      = (bf16)p0;
      lds_p[qloc * 32 + 16 + n] = (bf16)p1;
    }
    __syncthreads();  // single-wave WG: orders LDS store->load

    // ---- P (A-layout from LDS) x V (B from transposed V, contiguous) ----
    const bf16* pr = &lds_p[(lane & 15) * 32 + kbA];
    bf16x8 plo = *(const bf16x8*)(pr);
    bf16x8 phi = *(const bf16x8*)(pr + 8);
    bf16x16 ap = __builtin_shufflevector(plo, phi, 0, 1, 2, 3, 4, 5, 6, 7,
                                         8, 9, 10, 11, 12, 13, 14, 15);
#pragma unroll
    for (int t4 = 0; t4 < 4; ++t4) {
      const bf16* vrow = vp + (size_t)(t4 * 16 + n) * SV + kc + kb2;
      bf16x16 bv = *(const bf16x16*)vrow;  // pad keys pair with P==0
      o[t4] = wmma_bf16(ap, bv, o[t4]);
    }
    __syncthreads();
  }

  // ---- normalize + store ctx [B*S, D] bf16 ----
  const int bb = bh / Hn;
  const int h  = bh % Hn;
#pragma unroll
  for (int r = 0; r < 8; ++r) {
    const float inv  = 1.0f / l[r];
    const int   qrow = q0 + r + 8 * g;
    const size_t rowoff = ((size_t)bb * Sn + qrow) * Dn + h * HDn;
#pragma unroll
    for (int t4 = 0; t4 < 4; ++t4)
      ctx[rowoff + t4 * 16 + n] = (bf16)(o[t4][r] * inv);
  }
}

// ---------------------------------------------------------------------------
// Kernel 4: output projection  out[n,c] = sum_k ctx[n,k]*Wp[c,k] + bp[c] (f32)
// One wave -> 32 rows x 64 cols, same tiling as QKV GEMM.
// ---------------------------------------------------------------------------
__global__ __launch_bounds__(256) void swa_proj_gemm(
    const bf16* __restrict__ ctx, const bf16* __restrict__ wb,
    const float* __restrict__ bias, float* __restrict__ out) {
  const int lane = threadIdx.x & 31;
  const int unit = blockIdx.x * (blockDim.x >> 5) + (threadIdx.x >> 5);
  constexpr int COLG = Dn / 64;  // 16
  const int rt = unit / COLG;    // 32-row tile (Bn*Sn/32 = 128)
  const int cg = unit % COLG;
  const int n  = lane & 15;
  const int g  = lane >> 4;

  f32x8 acc[2][4] = {};
  for (int k0 = 0; k0 < Dn; k0 += 32) {
    bf16x16 a0 = load_a_tile(ctx, rt * 32, Dn, k0, lane);
    bf16x16 a1 = load_a_tile(ctx, rt * 32 + 16, Dn, k0, lane);
#pragma unroll
    for (int t = 0; t < 4; ++t) {
      const int c = cg * 64 + t * 16 + n;
      bf16x16 b = load_b_tile(wb + (size_t)c * Dn, k0, lane);
      acc[0][t] = wmma_bf16(a0, b, acc[0][t]);
      acc[1][t] = wmma_bf16(a1, b, acc[1][t]);
    }
  }

#pragma unroll
  for (int t = 0; t < 4; ++t) {
    const int c    = cg * 64 + t * 16 + n;
    const float bv = bias[c];
#pragma unroll
    for (int half = 0; half < 2; ++half) {
#pragma unroll
      for (int r = 0; r < 8; ++r) {
        const int grow = rt * 32 + half * 16 + r + 8 * g;
        out[(size_t)grow * Dn + c] = acc[half][t][r] + bv;
      }
    }
  }
}

// ---------------------------------------------------------------------------
// Host launcher
// ---------------------------------------------------------------------------
extern "C" void kernel_launch(void* const* d_in, const int* in_sizes, int n_in,
                              void* d_out, int out_size, void* d_ws,
                              size_t ws_size, hipStream_t stream) {
  const float* x      = (const float*)d_in[0];
  const float* qkv_w  = (const float*)d_in[1];
  const float* qkv_b  = (const float*)d_in[2];
  const float* proj_w = (const float*)d_in[3];
  const float* proj_b = (const float*)d_in[4];
  float*       out    = (float*)d_out;
  (void)in_sizes; (void)n_in; (void)out_size; (void)ws_size;

  // workspace layout (bf16, all regions 256B-aligned by construction)
  char* ws = (char*)d_ws;
  const size_t n_x   = (size_t)Bn * Sn * Dn;      // 4,194,304
  const size_t n_wq  = (size_t)D3 * Dn;           // 3,145,728
  const size_t n_wp  = (size_t)Dn * Dn;           // 1,048,576
  const size_t n_qk  = (size_t)Bn * Hn * Sn * HDn;
  const size_t n_vt  = (size_t)Bn * Hn * HDn * SV;

  bf16* xb    = (bf16*)(ws);
  bf16* wqkvb = xb + n_x;
  bf16* wprjb = wqkvb + n_wq;
  bf16* qB    = wprjb + n_wp;
  bf16* kB    = qB + n_qk;
  bf16* vtB   = kB + n_qk;
  bf16* ctx   = vtB + n_vt;   // total ~48.5 MB

  // 1) casts (4 elems/thread)
  swa_cast_bf16<<<2048, 256, 0, stream>>>(x, xb, (int)(n_x / 4));
  swa_cast_bf16<<<2048, 256, 0, stream>>>(qkv_w, wqkvb, (int)(n_wq / 4));
  swa_cast_bf16<<<1024, 256, 0, stream>>>(proj_w, wprjb, (int)(n_wp / 4));

  // 2) QKV GEMM: (B*S/32)*(3D/64) = 6144 waves, 8 waves/block
  swa_qkv_gemm<<<6144 / 8, 256, 0, stream>>>(xb, wqkvb, qkv_b, qB, kB, vtB);

  // 3) attention: one wave per (16-query tile, b*h)
  swa_attention<<<dim3(Sn / 16, Bn * Hn), 32, 0, stream>>>(qB, kB, vtB, ctx);

  // 4) projection: (B*S/32)*(D/64) = 2048 waves, 8 waves/block
  swa_proj_gemm<<<2048 / 8, 256, 0, stream>>>(ctx, wprjb, proj_b, out);
}
